// AttentionDecoder_77927886618807
// MI455X (gfx1250) — compile-verified, matrix operand order checked by reading, not run on hardware
//
#include <hip/hip_runtime.h>
#include <hip/hip_bf16.h>

typedef __attribute__((ext_vector_type(16))) _Float16 v16h;
typedef __attribute__((ext_vector_type(8)))  _Float16 v8h;
typedef __attribute__((ext_vector_type(8)))  float    v8f;

#define B_ 256
#define S_ 512
#define E_ 128
#define H_ 128

__device__ inline v8f wmma_f16(v16h a, v16h b, v8f c) {
  // D = A(16x32 f16) * B(32x16 f16) + C(16x16 f32)
  return __builtin_amdgcn_wmma_f32_16x16x32_f16(
      /*neg_a=*/false, a, /*neg_b=*/false, b,
      /*c_mod=*/(short)0, c, /*reuse_a=*/false, /*reuse_b=*/false);
}

__device__ inline v16h join8(v8h lo, v8h hi) {
  return __builtin_shufflevector(lo, hi, 0, 1, 2, 3, 4, 5, 6, 7,
                                 8, 9, 10, 11, 12, 13, 14, 15);
}

__device__ inline float sigmoidf_(float x) { return 1.0f / (1.0f + __expf(-x)); }

// ---------------------------------------------------------------------------
// Kernel 1: scores2 = encoded_hidden @ W2 + b2   ((B*S) x E) @ (E x E)
// W2 staged once per block into LDS in WMMA-fragment order:
//   [ntile(8)][kstep(4)][lane(32)][16 halves]  -> v16h per lane via 2x b128.
// ---------------------------------------------------------------------------
__global__ __launch_bounds__(256) void scores2_kernel(
    const float* __restrict__ enc, const float* __restrict__ W2,
    const float* __restrict__ b2, float* __restrict__ sc2) {
  __shared__ _Float16 w2f[8 * 4 * 32 * 16];     // 32 KB, fragment-swizzled
  const int tid = threadIdx.x;

  for (int i = tid; i < 8 * 4 * 32 * 16; i += 256) {
    const int j  = i & 15;
    const int ln = (i >> 4) & 31;
    const int ks = (i >> 9) & 3;
    const int nt = i >> 11;
    const int k   = ks * 32 + ((ln & 16) ? 8 : 0) + (j < 8 ? j : 8 + j);
    const int col = nt * 16 + (ln & 15);
    w2f[i] = (_Float16)W2[k * E_ + col];
  }
  __syncthreads();

  const int lane = tid & 31;
  const int wave = tid >> 5;                    // 0..7 -> n-tile
  const long m0 = (long)blockIdx.x * 16;        // row tile over B*S
  const int r  = lane & 15;
  const int kboff = (lane & 16) ? 8 : 0;
  const v8h* bfrag = (const v8h*)w2f;
  const float* arow = enc + (m0 + r) * E_;

  v8f acc = {};
#pragma unroll
  for (int ks = 0; ks < 4; ++ks) {
    const int kb = ks * 32 + kboff;
    v16h a;
#pragma unroll
    for (int j = 0; j < 8; ++j) {
      a[j]     = (_Float16)arow[kb + j];
      a[8 + j] = (_Float16)arow[kb + 16 + j];
    }
    const int fi = ((wave * 4 + ks) * 32 + lane) * 2;
    const v16h bm = join8(bfrag[fi], bfrag[fi + 1]);
    acc = wmma_f16(a, bm, acc);
  }
  const int col = wave * 16 + r;
  const int ro = (lane & 16) ? 8 : 0;
  const float bias = b2[col];
#pragma unroll
  for (int rr = 0; rr < 8; ++rr)
    sc2[(m0 + rr + ro) * E_ + col] = acc[rr] + bias;
}

// ---------------------------------------------------------------------------
// Kernel 2: full recurrent decoder. One workgroup per 16-batch tile.
// 512 threads = 16 wave32. h/c kept as row-major f16 (WMMA-A feed) with the
// c recurrence accumulated in f32; fragment-swizzled f16 weights in LDS.
// ---------------------------------------------------------------------------
__global__ __launch_bounds__(512) void decoder_kernel(
    const float* __restrict__ enc, const float* __restrict__ tgt,
    const float* __restrict__ W1, const float* __restrict__ b1,
    const float* __restrict__ W3, const float* __restrict__ b3,
    const float* __restrict__ Wih, const float* __restrict__ Whh,
    const float* __restrict__ bih, const float* __restrict__ bhh,
    const float* __restrict__ Wc, const float* __restrict__ bc,
    const float* __restrict__ Wf1, const float* __restrict__ bf1,
    const float* __restrict__ Wf2, const float* __restrict__ bf2,
    const float* __restrict__ sc2, float* __restrict__ out) {
  const int tid  = threadIdx.x;
  const int lane = tid & 31;
  const int wave = tid >> 5;                // 0..15
  const int b0   = blockIdx.x * 16;

  extern __shared__ char smem[];
  // W1 fragments: [ntile(8)][kstep(8)][lane(32)][16] = 32768 halves = 64 KB
  _Float16* W1f  = (_Float16*)smem;
  // Whh^T fragments: [ntile(32)][kstep(4)][lane(32)][16] = 65536 halves = 128 KB
  _Float16* Whf  = W1f + 8 * 8 * 32 * 16;
  float* cS   = (float*)(Whf + 32 * 4 * 32 * 16);   // 16*128 f32 (c recurrence)
  float* ctxS = cS + 16 * 128;                      // 16*128
  float* a1S  = ctxS + 16 * 128;                    // 16*128
  float* wS   = a1S + 16 * 128;                     // 16*512 (scores -> beta)
  float* gS   = wS + 16 * 512;                      // 16*512 (gates)
  float* yS   = gS + 16 * 512;                      // 16
  float* w3S  = yS + 16;                            // 128
  _Float16* h16 = (_Float16*)(w3S + 128);           // 16*128 f16 (WMMA A feed)
  _Float16* c16 = h16 + 16 * 128;                   // 16*128 f16 (WMMA A feed)

  // ---- one-time init: stage weights as fragment-swizzled f16, zero state ----
  for (int i = tid; i < 8 * 8 * 32 * 16; i += 512) {      // W1 (256 x 128)
    const int j  = i & 15;
    const int ln = (i >> 4) & 31;
    const int ks = (i >> 9) & 7;
    const int nt = i >> 12;
    const int k   = ks * 32 + ((ln & 16) ? 8 : 0) + (j < 8 ? j : 8 + j);
    const int col = nt * 16 + (ln & 15);
    W1f[i] = (_Float16)W1[k * E_ + col];
  }
  for (int i = tid; i < 32 * 4 * 32 * 16; i += 512) {     // Whh^T (128 x 512)
    const int j  = i & 15;
    const int ln = (i >> 4) & 31;
    const int ks = (i >> 9) & 3;
    const int nt = i >> 11;
    const int k   = ks * 32 + ((ln & 16) ? 8 : 0) + (j < 8 ? j : 8 + j);
    const int col = nt * 16 + (ln & 15);
    Whf[i] = (_Float16)Whh[col * H_ + k];                 // B[k][col] = Whh[col][k]
  }
  for (int i = tid; i < 16 * 128; i += 512) {
    cS[i] = 0.f; ctxS[i] = 0.f;
    h16[i] = (_Float16)0.f; c16[i] = (_Float16)0.f;
  }
  for (int i = tid; i < 128; i += 512) w3S[i] = W3[i];
  __syncthreads();

  const int r15   = lane & 15;
  const int kboff = (lane & 16) ? 8 : 0;
  const int ro    = (lane & 16) ? 8 : 0;
  const v8h* w1frag = (const v8h*)W1f;
  const v8h* whfrag = (const v8h*)Whf;

  for (int t = 0; t < S_; ++t) {
    // ---- Phase A: a1 = [h|c] @ W1 + b1  (waves 0..7, one 16-col tile each)
    if (wave < 8) {
      v8f acc = {};
#pragma unroll
      for (int ks = 0; ks < 8; ++ks) {
        const _Float16* src = (ks < 4) ? h16 : c16;       // k<128 -> h else c
        const int kb = (ks & 3) * 32 + kboff;
        const v8h* av = (const v8h*)(src + r15 * H_ + kb);
        const v16h a = join8(av[0], av[2]);               // kb..kb+7, kb+16..kb+23
        const int fi = ((wave * 8 + ks) * 32 + lane) * 2;
        const v16h bm = join8(w1frag[fi], w1frag[fi + 1]);
        acc = wmma_f16(a, bm, acc);
      }
      const int col = wave * 16 + r15;
      const float bias = b1[col];
#pragma unroll
      for (int rr = 0; rr < 8; ++rr)
        a1S[(rr + ro) * E_ + col] = acc[rr] + bias;
    }
    __syncthreads();

    // ---- Phase B: w[b][s] = b3 + sum_e tanh(a1[b][e] + sc2[b][s][e]) * W3[e]
    {
      const int s = tid;                      // 512 threads == S_
      const float bias3 = b3[0];
      for (int b = 0; b < 16; ++b) {
        const float* row = sc2 + ((size_t)(b0 + b) * S_ + s) * E_;
        const float* a1r = a1S + b * E_;
        float acc = bias3;
#pragma unroll 4
        for (int e = 0; e < E_; ++e)
          acc += tanhf(a1r[e] + row[e]) * w3S[e];
        wS[b * S_ + s] = acc;
      }
    }
    __syncthreads();

    // ---- Phase C: softmax over S, one wave per batch row
    {
      float* row = wS + wave * S_;
      float m = -3.4e38f;
      for (int s = lane; s < S_; s += 32) m = fmaxf(m, row[s]);
#pragma unroll
      for (int off = 16; off > 0; off >>= 1) m = fmaxf(m, __shfl_xor(m, off, 32));
      float sum = 0.f;
      for (int s = lane; s < S_; s += 32) {
        const float ev = __expf(row[s] - m);
        row[s] = ev; sum += ev;
      }
#pragma unroll
      for (int off = 16; off > 0; off >>= 1) sum += __shfl_xor(sum, off, 32);
      const float inv = 1.f / sum;
      for (int s = lane; s < S_; s += 32) row[s] *= inv;
    }
    __syncthreads();

    // ---- Phase D: ctx[b][e] = sum_s beta[b][s] * enc[b][s][e]
    {
      const int e  = tid & 127;
      const int bg = tid >> 7;                // 0..3
#pragma unroll
      for (int rep = 0; rep < 4; ++rep) {
        const int b = rep * 4 + bg;
        const float* base = enc + (size_t)(b0 + b) * S_ * E_ + e;
        const float* beta = wS + b * S_;
        float acc = 0.f;
        for (int s = 0; s < S_; ++s) acc += beta[s] * base[(size_t)s * E_];
        ctxS[b * E_ + e] = acc;
      }
    }
    __syncthreads();

    if (t < S_ - 1) {
      // ---- Phase E: y[b] = [tgt_t, ctx] . Wc + bc
      if (tid < 16) {
        float acc = tgt[(size_t)(b0 + tid) * S_ + t] * Wc[0] + bc[0];
        const float* cr = ctxS + tid * E_;
        for (int e = 0; e < E_; ++e) acc += cr[e] * Wc[1 + e];
        yS[tid] = acc;
      }
      __syncthreads();

      // ---- Phase F: gates = h @ Whh^T + y*Wih + bih + bhh (16 waves x 2 tiles)
#pragma unroll
      for (int nt = 0; nt < 2; ++nt) {
        const int ntile = wave * 2 + nt;
        v8f acc = {};
#pragma unroll
        for (int ks = 0; ks < 4; ++ks) {
          const int kb = ks * 32 + kboff;
          const v8h* av = (const v8h*)(h16 + r15 * H_ + kb);
          const v16h a = join8(av[0], av[2]);
          const int fi = ((ntile * 4 + ks) * 32 + lane) * 2;
          const v16h bm = join8(whfrag[fi], whfrag[fi + 1]);
          acc = wmma_f16(a, bm, acc);
        }
        const int col = ntile * 16 + r15;
        const float wi = Wih[col], bi = bih[col] + bhh[col];
#pragma unroll
        for (int rr = 0; rr < 8; ++rr) {
          const int b = rr + ro;
          gS[b * 512 + col] = acc[rr] + yS[b] * wi + bi;
        }
      }
      __syncthreads();

      // ---- Phase G: LSTM elementwise update (gate order i,f,g,o)
#pragma unroll
      for (int rep = 0; rep < 4; ++rep) {
        const int idx = rep * 512 + tid;
        const int b = idx >> 7, e = idx & 127;
        const float* g = gS + b * 512;
        const float ig = sigmoidf_(g[e]);
        const float fg = sigmoidf_(g[128 + e]);
        const float gg = tanhf(g[256 + e]);
        const float og = sigmoidf_(g[384 + e]);
        const float cn = fg * cS[b * H_ + e] + ig * gg;
        const float hn = og * tanhf(cn);
        cS[b * H_ + e]  = cn;                  // f32 recurrence state
        c16[b * H_ + e] = (_Float16)cn;        // WMMA A feed
        h16[b * H_ + e] = (_Float16)hn;        // WMMA A feed
      }
      __syncthreads();
    }
  }

  // ---- Final: out = ([h|ctx] @ Wf1 + bf1) @ Wf2 + bf2   (reuse a1S)
  if (wave < 8) {
    const int col = wave * 16 + r15;
    v8f acc = {};
#pragma unroll
    for (int ks = 0; ks < 8; ++ks) {
      const int kb = (ks & 3) * 32 + kboff;
      const int kglob = ks * 32 + kboff;
      v16h a, bm;
      if (ks < 4) {                              // h half: direct f16
        const v8h* av = (const v8h*)(h16 + r15 * H_ + kb);
        a = join8(av[0], av[2]);
      } else {                                   // ctx half: convert f32
#pragma unroll
        for (int j = 0; j < 8; ++j) {
          a[j]     = (_Float16)ctxS[r15 * H_ + kb + j];
          a[8 + j] = (_Float16)ctxS[r15 * H_ + kb + 16 + j];
        }
      }
#pragma unroll
      for (int j = 0; j < 8; ++j) {
        bm[j]     = (_Float16)Wf1[(kglob + j) * H_ + col];
        bm[8 + j] = (_Float16)Wf1[(kglob + 16 + j) * H_ + col];
      }
      acc = wmma_f16(a, bm, acc);
    }
    const float bias = bf1[col];
#pragma unroll
    for (int rr = 0; rr < 8; ++rr)
      a1S[(rr + ro) * H_ + col] = acc[rr] + bias;
  }
  __syncthreads();
  if (tid < 16) {
    float acc = bf2[0];
    const float* fr = a1S + tid * H_;
    for (int e = 0; e < H_; ++e) acc += fr[e] * Wf2[e];
    out[b0 + tid] = acc;
  }
}

// ---------------------------------------------------------------------------
extern "C" void kernel_launch(void* const* d_in, const int* in_sizes, int n_in,
                              void* d_out, int out_size, void* d_ws, size_t ws_size,
                              hipStream_t stream) {
  (void)in_sizes; (void)n_in; (void)out_size; (void)ws_size;
  const float* enc = (const float*)d_in[0];
  const float* tgt = (const float*)d_in[1];
  const float* W1  = (const float*)d_in[2];
  const float* b1  = (const float*)d_in[3];
  const float* W2  = (const float*)d_in[4];
  const float* b2  = (const float*)d_in[5];
  const float* W3  = (const float*)d_in[6];
  const float* b3  = (const float*)d_in[7];
  const float* Wih = (const float*)d_in[8];
  const float* Whh = (const float*)d_in[9];
  const float* bih = (const float*)d_in[10];
  const float* bhh = (const float*)d_in[11];
  const float* Wc  = (const float*)d_in[12];
  const float* bc  = (const float*)d_in[13];
  const float* Wf1 = (const float*)d_in[14];
  const float* bf1 = (const float*)d_in[15];
  const float* Wf2 = (const float*)d_in[16];
  const float* bf2 = (const float*)d_in[17];

  float* sc2 = (float*)d_ws;                 // B*S*E f32 = 64 MB scratch
  float* out = (float*)d_out;                // (B,1) f32

  // Precompute time-invariant scores2 = enc @ W2 + b2
  scores2_kernel<<<(B_ * S_) / 16, 256, 0, stream>>>(enc, W2, b2, sc2);

  // Recurrent decoder: 16 workgroups x 512 threads, ~289 KB dynamic LDS each
  const size_t lds = (size_t)(8 * 8 * 32 * 16 + 32 * 4 * 32 * 16) * sizeof(_Float16) +
                     (size_t)(3 * 16 * 128 + 2 * 16 * 512 + 16 + 128) * sizeof(float) +
                     (size_t)(2 * 16 * 128) * sizeof(_Float16);
  decoder_kernel<<<B_ / 16, 512, lds, stream>>>(
      enc, tgt, W1, b1, W3, b3, Wih, Whh, bih, bhh, Wc, bc,
      Wf1, bf1, Wf2, bf2, sc2, out);
}
